// ViT_72395968741404
// MI455X (gfx1250) — matrix operand order repaired and checked by reading, not compiled, not run on hardware
//
#include <hip/hip_runtime.h>

typedef unsigned short u16;
typedef unsigned int u32;
typedef __attribute__((ext_vector_type(16))) __bf16 v16bf;
typedef __attribute__((ext_vector_type(8)))  float  v8f;

// ---------- ViT-Base/16 config ----------
#define KB   32          // batch
#define KD   768         // model dim
#define KL   12          // layers
#define KNH  12          // heads
#define KHS  64          // head size
#define KHID 3072        // mlp hidden
#define KNP  196         // patches
#define KP   197         // tokens (patches + cls)
#define KM   (KB*KP)     // 6304 token rows
#define KPP  224         // padded attention row (multiple of 32)

// ---------- helpers ----------
__device__ __forceinline__ u16 f2bf(float f) {
    u32 u = __float_as_uint(f);
    if ((u & 0x7fffffffu) > 0x7f800000u) return 0x7FC0;      // NaN
    u32 r = (u + 0x7FFFu + ((u >> 16) & 1u)) >> 16;          // RNE
    return (u16)r;
}
__device__ __forceinline__ float gelu_f(float x) {
    float x3 = x * x * x;
    return 0.5f * x * (1.0f + tanhf(0.7978845608028654f * (x + 0.044715f * x3)));
}

union FragU { uint4 u[2]; v16bf v; };

// CDNA5 async global->LDS copy (16B per lane), tracked by ASYNCcnt.
// lds generic pointer's low 32 bits == LDS byte offset (aperture in high dword).
__device__ __forceinline__ void async_cp16(void* lds_dst, const void* gsrc) {
    u32 loff = (u32)(unsigned long long)lds_dst;
    asm volatile("global_load_async_to_lds_b128 %0, %1, off"
                 :: "v"(loff), "v"(gsrc) : "memory");
}
// async loads complete in-order: waiting <=N leaves the newest N in flight
__device__ __forceinline__ void async_wait0() {
    asm volatile("s_wait_asynccnt 0x0" ::: "memory");
}
__device__ __forceinline__ void async_wait2() {
    asm volatile("s_wait_asynccnt 0x2" ::: "memory");
}
__device__ __forceinline__ void async_wait4() {
    asm volatile("s_wait_asynccnt 0x4" ::: "memory");
}

// ---------- fp32 -> bf16 convert ----------
__global__ void cvt_bf16(const float* __restrict__ in, u16* __restrict__ out, int n) {
    int i = blockIdx.x * 256 + threadIdx.x;
    if (i < n) out[i] = f2bf(in[i]);
}

// per-z [K][N] fp32 -> [N][K] bf16
__global__ void cvtT_bf16(const float* __restrict__ in, u16* __restrict__ out, int K, int N) {
    long long base = (long long)blockIdx.z * K * N;
    int idx = blockIdx.x * 256 + threadIdx.x;
    if (idx >= K * N) return;
    int k = idx % K, n = idx / K;
    out[base + idx] = f2bf(in[base + (long long)k * N + n]);
}

// ---------- patchify: x[B,3,224,224] -> patches[B*196][768] bf16 ----------
__global__ void patchify_kernel(const float* __restrict__ x, u16* __restrict__ p, int total) {
    int idx = blockIdx.x * 256 + threadIdx.x;
    if (idx >= total) return;
    int k = idx % KD;
    int t = idx / KD;
    int pch = t % KNP, b = t / KNP;
    int c = k >> 8, rr = (k >> 4) & 15, cc = k & 15;
    int gy = pch / 14, gx = pch % 14;
    long long src = (((long long)(b * 3 + c) * 224) + gy * 16 + rr) * 224 + gx * 16 + cc;
    p[idx] = f2bf(x[src]);
}

// ---------- cls row + positional embedding ----------
__global__ void embed_kernel(float* __restrict__ h, const float* __restrict__ cls,
                             const float* __restrict__ pos, int total) {
    int idx = blockIdx.x * 256 + threadIdx.x;
    if (idx >= total) return;
    int d = idx % KD;
    int t = (idx / KD) % KP;
    if (t == 0) h[idx] = cls[d] + pos[d];
    else        h[idx] += pos[t * KD + d];
}

// ---------- layernorm: h fp32 -> bf16 (one wave per row) ----------
__global__ void layernorm_bf16(const float* __restrict__ X, const float* __restrict__ w,
                               const float* __restrict__ b, u16* __restrict__ Y, int M) {
    int row  = blockIdx.x * 8 + (threadIdx.x >> 5);
    int lane = threadIdx.x & 31;
    if (row >= M) return;
    const float* x = X + (long long)row * KD;
    float s = 0.f, s2 = 0.f;
    for (int i = lane; i < KD; i += 32) { float v = x[i]; s += v; s2 += v * v; }
    for (int o = 16; o > 0; o >>= 1) { s += __shfl_xor(s, o, 32); s2 += __shfl_xor(s2, o, 32); }
    float mean = s * (1.f / KD);
    float var  = s2 * (1.f / KD) - mean * mean;
    float inv  = rsqrtf(var + 1e-5f);
    u16* y = Y + (long long)row * KD;
    for (int i = lane; i < KD; i += 32) y[i] = f2bf((x[i] - mean) * inv * w[i] + b[i]);
}

// ---------- softmax rows of 224 (197 valid), fp32 in -> bf16 out ----------
__global__ void softmax_bf16(const float* __restrict__ S, u16* __restrict__ P,
                             int rows, float scale) {
    int row  = blockIdx.x * 8 + (threadIdx.x >> 5);
    int lane = threadIdx.x & 31;
    if (row >= rows) return;
    const float* s = S + (long long)row * KPP;
    u16* p = P + (long long)row * KPP;
    float vals[7];
    float mx = -1e30f;
    #pragma unroll
    for (int i = 0; i < 7; ++i) {
        int c = lane + i * 32;
        float v = (c < KP) ? s[c] * scale : -1e30f;
        vals[i] = v; mx = fmaxf(mx, v);
    }
    for (int o = 16; o > 0; o >>= 1) mx = fmaxf(mx, __shfl_xor(mx, o, 32));
    float sum = 0.f;
    #pragma unroll
    for (int i = 0; i < 7; ++i) {
        float e = (vals[i] > -1e29f) ? __expf(vals[i] - mx) : 0.f;
        vals[i] = e; sum += e;
    }
    for (int o = 16; o > 0; o >>= 1) sum += __shfl_xor(sum, o, 32);
    float r = 1.f / sum;
    #pragma unroll
    for (int i = 0; i < 7; ++i) p[lane + i * 32] = f2bf(vals[i] * r);
}

// ---------- WMMA GEMM: block tile 128x128, wave tile 32x64, BK=32 ----------
// 3-stage LDS ring, async global->LDS staging with relaxed ASYNCcnt waits
// A: bf16 [M][K] (lda); B: !BKMAJ [N][K] (ldb) | BKMAJ [K][N] (ldb, LDS transpose)
// z offsets: off = (z/zdiv)*S1 + (z%zdiv)*S2
// EPI: 0 none | 1 +bias | 2 +bias+resid | 3 gelu(+bias)
template<bool BKMAJ, int EPI, bool OBF16>
__global__ __launch_bounds__(256)
void gemm_wmma(const u16* __restrict__ Ag, long long aS1, long long aS2, int lda,
               const u16* __restrict__ Bg, long long bS1, long long bS2, int ldb,
               void* __restrict__ Cg, long long cS1, long long cS2, int ldc,
               const float* __restrict__ bias,
               const float* __restrict__ resid, long long rS1, long long rS2, int ldr,
               int M, int N, int K, int zdiv) {
    __shared__ __align__(16) u16 As[3][128][40];
    __shared__ __align__(16) u16 Bs[3][128][40];

    int z   = blockIdx.z;
    int zhi = z / zdiv, zlo = z % zdiv;
    const u16* A = Ag + (long long)zhi * aS1 + (long long)zlo * aS2;
    const u16* B = Bg + (long long)zhi * bS1 + (long long)zlo * bS2;
    long long cBase = (long long)zhi * cS1 + (long long)zlo * cS2;
    long long rBase = (long long)zhi * rS1 + (long long)zlo * rS2;

    int m0 = blockIdx.x * 128;
    int n0 = blockIdx.y * 128;
    int tid = threadIdx.x;
    int w = tid >> 5, lane = tid & 31;
    int wm = (w >> 1) * 32;     // 0,32,64,96
    int wn = (w & 1) * 64;      // 0,64
    int lm = lane & 15, lh = lane >> 4;

    v8f acc[2][4] = {};

    auto stage = [&](int buf, int k0) {
        // A tile 128x32: 512 16B-chunks, 2 per thread (async)
        #pragma unroll
        for (int i = 0; i < 2; ++i) {
            int id  = tid + i * 256;
            int row = id >> 2, c8 = (id & 3) * 8;
            int gr  = m0 + row; if (gr >= M) gr = M - 1;
            async_cp16(&As[buf][row][c8], A + (long long)gr * lda + k0 + c8);
        }
        if (!BKMAJ) {
            // B tile [N][K] 128x32 (async)
            #pragma unroll
            for (int i = 0; i < 2; ++i) {
                int id  = tid + i * 256;
                int row = id >> 2, c8 = (id & 3) * 8;
                int gr  = n0 + row; if (gr >= N) gr = N - 1;
                async_cp16(&Bs[buf][row][c8], B + (long long)gr * ldb + k0 + c8);
            }
        } else {
            // B tile [K][N] 32x128, transpose into LDS (sync loads, DScnt-covered)
            #pragma unroll
            for (int i = 0; i < 2; ++i) {
                int id = tid + i * 256;
                int kr = id >> 4, c8 = (id & 15) * 8;
                uint4 v = *(const uint4*)(B + (long long)(k0 + kr) * ldb + n0 + c8);
                const u16* pv = (const u16*)&v;
                #pragma unroll
                for (int j = 0; j < 8; ++j) Bs[buf][c8 + j][kr] = pv[j];
            }
        }
    };

    // prologue: launch stages 0 and 1
    stage(0, 0);
    if (32 < K) stage(1, 32);

    int buf = 0;
    for (int k0 = 0; k0 < K; k0 += 32) {
        int nk = k0 + 32;
        // stage for k0 resident (in-order completion); stage for nk may stay in flight
        if (nk < K) { if (BKMAJ) async_wait2(); else async_wait4(); }
        else        async_wait0();
        __syncthreads();   // everyone has stage k0; buffer (buf+2)%3 free to refill

        int k2 = k0 + 64;
        if (k2 < K) {
            int b2 = buf + 2; if (b2 >= 3) b2 -= 3;
            stage(b2, k2);   // async prefetch two stages ahead
        }

        // fragments per ISA layout:
        // A (16x32): lane half h -> K chunks [8h..8h+7] and [16+8h..16+8h+7]
        // B (32x16): lane half h -> K chunks [8h..8h+7] and [16+8h..16+8h+7]
        FragU fa[2], fb[4];
        #pragma unroll
        for (int i = 0; i < 2; ++i) {
            fa[i].u[0] = *(const uint4*)&As[buf][wm + i * 16 + lm][8 * lh];
            fa[i].u[1] = *(const uint4*)&As[buf][wm + i * 16 + lm][16 + 8 * lh];
        }
        #pragma unroll
        for (int j = 0; j < 4; ++j) {
            fb[j].u[0] = *(const uint4*)&Bs[buf][wn + j * 16 + lm][8 * lh];
            fb[j].u[1] = *(const uint4*)&Bs[buf][wn + j * 16 + lm][16 + 8 * lh];
        }
        #pragma unroll
        for (int i = 0; i < 2; ++i)
            #pragma unroll
            for (int j = 0; j < 4; ++j)
                acc[i][j] = __builtin_amdgcn_wmma_f32_16x16x32_bf16(
                    false, fa[i].v, false, fb[j].v, (short)0, acc[i][j], false, false);

        buf = (buf + 1 == 3) ? 0 : buf + 1;
    }

    // epilogue: C/D layout -> VGPR r, lane: row = r + 8*lh, col = lane&15
    float* Cf = (float*)Cg;
    u16*   Ch = (u16*)Cg;
    #pragma unroll
    for (int i = 0; i < 2; ++i)
        #pragma unroll
        for (int j = 0; j < 4; ++j)
            #pragma unroll
            for (int r = 0; r < 8; ++r) {
                int row = m0 + wm + i * 16 + 8 * lh + r;
                int col = n0 + wn + j * 16 + lm;
                if (row < M && col < N) {
                    float v = acc[i][j][r];
                    if (EPI >= 1 && bias) v += bias[col];
                    if (EPI == 2 && resid) v += resid[rBase + (long long)row * ldr + col];
                    if (EPI == 3) v = gelu_f(v);
                    long long off = cBase + (long long)row * ldc + col;
                    if (OBF16) Ch[off] = f2bf(v);
                    else       Cf[off] = v;
                }
            }
}

// ---------- classifier: out[b][n] = h[b][0][:] . W[:,n] + bias[n] ----------
__global__ void cls_kernel(const float* __restrict__ h, const float* __restrict__ W,
                           const float* __restrict__ bias, float* __restrict__ out, int NC) {
    __shared__ float red[256];
    int b = blockIdx.x, t = threadIdx.x;
    for (int n = 0; n < NC; ++n) {
        float s = 0.f;
        for (int i = t; i < KD; i += 256)
            s += h[(long long)b * KP * KD + i] * W[i * NC + n];
        red[t] = s; __syncthreads();
        for (int st = 128; st > 0; st >>= 1) { if (t < st) red[t] += red[t + st]; __syncthreads(); }
        if (t == 0) out[b * NC + n] = red[0] + bias[n];
        __syncthreads();
    }
}

// ====================================================================
extern "C" void kernel_launch(void* const* d_in, const int* in_sizes, int n_in,
                              void* d_out, int out_size, void* d_ws, size_t ws_size,
                              hipStream_t stream) {
    const float* x      = (const float*)d_in[0];
    const float* conv_w = (const float*)d_in[1];
    const float* conv_b = (const float*)d_in[2];
    const float* cls_tk = (const float*)d_in[3];
    const float* pos    = (const float*)d_in[4];
    const float* ln1w   = (const float*)d_in[5];
    const float* ln1b   = (const float*)d_in[6];
    const float* wq = (const float*)d_in[7];  const float* bq = (const float*)d_in[8];
    const float* wk = (const float*)d_in[9];  const float* bk = (const float*)d_in[10];
    const float* wv = (const float*)d_in[11]; const float* bv = (const float*)d_in[12];
    const float* wo = (const float*)d_in[13]; const float* bo = (const float*)d_in[14];
    const float* ln2w = (const float*)d_in[15]; const float* ln2b = (const float*)d_in[16];
    const float* w1 = (const float*)d_in[17]; const float* b1 = (const float*)d_in[18];
    const float* w2 = (const float*)d_in[19]; const float* b2 = (const float*)d_in[20];
    const float* clsw = (const float*)d_in[21]; const float* clsb = (const float*)d_in[22];
    const int NC = in_sizes[21] / KD;

    // ---- workspace carve ----
    char* ws = (char*)d_ws;
    size_t off = 0;
    auto alloc = [&](size_t bytes) -> char* {
        off = (off + 255) & ~(size_t)255;
        char* p = ws + off; off += bytes; return p;
    };
    u16* convW = (u16*)alloc((size_t)KD * KD * 2);
    u16* wqT = (u16*)alloc((size_t)KL * KD * KD * 2);
    u16* wkT = (u16*)alloc((size_t)KL * KD * KD * 2);
    u16* wvT = (u16*)alloc((size_t)KL * KD * KD * 2);
    u16* woT = (u16*)alloc((size_t)KL * KD * KD * 2);
    u16* w1T = (u16*)alloc((size_t)KL * KD * KHID * 2);
    u16* w2T = (u16*)alloc((size_t)KL * KD * KHID * 2);
    u16* patches = (u16*)alloc((size_t)KB * KNP * KD * 2);
    float* h  = (float*)alloc((size_t)KM * KD * 4);
    u16* hn   = (u16*)alloc((size_t)KM * KD * 2);
    u16* qb   = (u16*)alloc((size_t)KM * KD * 2);
    u16* kb   = (u16*)alloc((size_t)KM * KD * 2);
    u16* vb   = (u16*)alloc((size_t)(KM + 32) * KD * 2);   // pad rows for K=224 overread
    float* sc = (float*)alloc((size_t)KB * KNH * KP * KPP * 4);
    u16* pr   = (u16*)alloc((size_t)KB * KNH * KP * KPP * 2);
    u16* ob   = (u16*)alloc((size_t)KM * KD * 2);
    u16* ub   = (u16*)alloc((size_t)KM * KHID * 2);
    (void)ws_size;

    // ---- weight conversion ----
    cvt_bf16<<<(KD * KD + 255) / 256, 256, 0, stream>>>(conv_w, convW, KD * KD);
    {
        dim3 g((KD * KD + 255) / 256, 1, KL);
        cvtT_bf16<<<g, 256, 0, stream>>>(wq, wqT, KD, KD);
        cvtT_bf16<<<g, 256, 0, stream>>>(wk, wkT, KD, KD);
        cvtT_bf16<<<g, 256, 0, stream>>>(wv, wvT, KD, KD);
        cvtT_bf16<<<g, 256, 0, stream>>>(wo, woT, KD, KD);
        dim3 g1((KD * KHID + 255) / 256, 1, KL);
        cvtT_bf16<<<g1, 256, 0, stream>>>(w1, w1T, KD, KHID);   // [768][3072] -> [3072][768]
        cvtT_bf16<<<g1, 256, 0, stream>>>(w2, w2T, KHID, KD);   // [3072][768] -> [768][3072]
    }

    // ---- patchify + embed ----
    {
        int tot = KB * KNP * KD;
        patchify_kernel<<<(tot + 255) / 256, 256, 0, stream>>>(x, patches, tot);
        dim3 g((KNP + 127) / 128, KD / 128, KB);
        gemm_wmma<false, 1, false><<<g, 256, 0, stream>>>(
            patches, (long long)KNP * KD, 0, KD,
            convW, 0, 0, KD,
            (void*)(h + KD), (long long)KP * KD, 0, KD,   // skip cls row per batch
            conv_b, nullptr, 0, 0, 0,
            KNP, KD, KD, 1);
        int tot2 = KB * KP * KD;
        embed_kernel<<<(tot2 + 255) / 256, 256, 0, stream>>>(h, cls_tk, pos, tot2);
    }

    const int rowsLN  = (KM + 7) / 8;
    const int rowsSM  = (KB * KNH * KP + 7) / 8;
    dim3 gTok((KM + 127) / 128, KD / 128, 1);       // 50 x 6
    dim3 gMlp1((KM + 127) / 128, KHID / 128, 1);    // 50 x 24
    dim3 gScr((KP + 127) / 128, (KP + 127) / 128, KB * KNH);  // 2 x 2 x 384
    dim3 gAV((KP + 127) / 128, 1, KB * KNH);                  // 2 x 1 x 384

    for (int l = 0; l < KL; ++l) {
        const u16* wqTl = wqT + (size_t)l * KD * KD;
        const u16* wkTl = wkT + (size_t)l * KD * KD;
        const u16* wvTl = wvT + (size_t)l * KD * KD;
        const u16* woTl = woT + (size_t)l * KD * KD;
        const u16* w1Tl = w1T + (size_t)l * KD * KHID;
        const u16* w2Tl = w2T + (size_t)l * KD * KHID;

        // LN1
        layernorm_bf16<<<rowsLN, 256, 0, stream>>>(h, ln1w + l * KD, ln1b + l * KD, hn, KM);
        // Q,K,V projections -> bf16
        gemm_wmma<false, 1, true><<<gTok, 256, 0, stream>>>(
            hn, 0, 0, KD, wqTl, 0, 0, KD, (void*)qb, 0, 0, KD,
            bq + l * KD, nullptr, 0, 0, 0, KM, KD, KD, 1);
        gemm_wmma<false, 1, true><<<gTok, 256, 0, stream>>>(
            hn, 0, 0, KD, wkTl, 0, 0, KD, (void*)kb, 0, 0, KD,
            bk + l * KD, nullptr, 0, 0, 0, KM, KD, KD, 1);
        gemm_wmma<false, 1, true><<<gTok, 256, 0, stream>>>(
            hn, 0, 0, KD, wvTl, 0, 0, KD, (void*)vb, 0, 0, KD,
            bv + l * KD, nullptr, 0, 0, 0, KM, KD, KD, 1);
        // scores = Q @ K^T  (per batch,head)
        gemm_wmma<false, 0, false><<<gScr, 256, 0, stream>>>(
            qb, (long long)KP * KD, KHS, KD,
            kb, (long long)KP * KD, KHS, KD,
            (void*)sc, (long long)KNH * KP * KPP, (long long)KP * KPP, KPP,
            nullptr, nullptr, 0, 0, 0, KP, KP, KHS, KNH);
        // softmax(scale) -> bf16 probs, zero-padded to 224
        softmax_bf16<<<rowsSM, 256, 0, stream>>>(sc, pr, KB * KNH * KP, 0.125f);
        // o = probs @ V   (B is [K][N] row-major -> BKMAJ path)
        gemm_wmma<true, 0, true><<<gAV, 256, 0, stream>>>(
            pr, (long long)KNH * KP * KPP, (long long)KP * KPP, KPP,
            vb, (long long)KP * KD, KHS, KD,
            (void*)ob, (long long)KP * KD, KHS, KD,
            nullptr, nullptr, 0, 0, 0, KP, KHS, KPP, KNH);
        // h += o @ Wo + bo
        gemm_wmma<false, 2, false><<<gTok, 256, 0, stream>>>(
            ob, 0, 0, KD, woTl, 0, 0, KD, (void*)h, 0, 0, KD,
            bo + l * KD, h, 0, 0, KD, KM, KD, KD, 1);
        // LN2
        layernorm_bf16<<<rowsLN, 256, 0, stream>>>(h, ln2w + l * KD, ln2b + l * KD, hn, KM);
        // u = gelu(hn @ W1 + b1)
        gemm_wmma<false, 3, true><<<gMlp1, 256, 0, stream>>>(
            hn, 0, 0, KD, w1Tl, 0, 0, KD, (void*)ub, 0, 0, KHID,
            b1 + l * KHID, nullptr, 0, 0, 0, KM, KHID, KD, 1);
        // h += u @ W2 + b2
        gemm_wmma<false, 2, false><<<gTok, 256, 0, stream>>>(
            ub, 0, 0, KHID, w2Tl, 0, 0, KHID, (void*)h, 0, 0, KD,
            b2 + l * KD, h, 0, 0, KD, KM, KD, KHID, 1);
    }

    // classifier head
    cls_kernel<<<KB, 256, 0, stream>>>(h, clsw, clsb, (float*)d_out, NC);
    (void)out_size; (void)n_in;
}
